// MPNN_42142219108493
// MI455X (gfx1250) — matrix operand order repaired and checked by reading, not compiled, not run on hardware
//
#include <hip/hip_runtime.h>
#include <hip/hip_bf16.h>
#include <stdint.h>
#include <stddef.h>

// ---------------------------------------------------------------------------
// MPNN (NNConv + GRU + Set2Set) for MI455X / gfx1250, wave32 + WMMA.
//
//  * W_e (E x 64 x 64) materialized ONCE in f16 (134 MB -> fits 192 MB L2),
//    produced by a WMMA GEMM [16384,128]x[128,4096] (~17 GFLOP).
//  * GEMM: block = 4 waves = 64(M) x 64(N) tile; B panel staged in LDS
//    (async global->LDS when available, padded rows -> conflict-free
//    16-bit LDS reads), K fully unrolled, A fragments preloaded in one clause,
//    all 4 B fragments of a k-step loaded before the 4 WMMAs.
//  * GRU gate GEMMs (8192x192x64) use the same kernel with K=64.
//  * Per-edge message = wave-per-edge f16 dot rows + f32 atomics.
//  * Set2Set: small VALU kernels, ordered-uint atomicMax for segment_max.
// ---------------------------------------------------------------------------

typedef __attribute__((ext_vector_type(16))) _Float16 v16h;
typedef __attribute__((ext_vector_type(8)))  _Float16 v8h;
typedef __attribute__((ext_vector_type(2)))  _Float16 v2h;
typedef __attribute__((ext_vector_type(8)))  float    v8f;
typedef int v4i_t __attribute__((vector_size(16)));   // matches async-LDS builtin param

#define HD 64   // hidden size H

#if defined(__AMDGCN__) && __has_builtin(__builtin_amdgcn_global_load_async_to_lds_b128)
#define USE_ASYNC_LDS 1
#else
#define USE_ASYNC_LDS 0
#endif

// ---------------------------------------------------------------------------
// WMMA GEMM:  C[M,N] = A[M,K](f16,row-major) * B[K,N](f16,row-major) + bias.
// Block = 128 threads = 4 waves. Block tile = 64(M) x 64(N). K is a compile-
// time constant (multiple of 32). M % 64 == 0, N % 64 == 0.
// ---------------------------------------------------------------------------
template<int K, bool HALF_OUT>
__global__ __launch_bounds__(128)
void wmma_gemm_kernel(const _Float16* __restrict__ A,
                      const _Float16* __restrict__ B,
                      const float*    __restrict__ bias,
                      void*           __restrict__ C,
                      int M, int N)
{
    constexpr int LDB = 72;          // padded f16 row stride (bank-conflict-free)
    constexpr int KS  = K / 32;      // number of k-steps
    __shared__ _Float16 Bs[K * LDB];

    const int nsup = N >> 6;                         // N / 64
    const int bm   = blockIdx.x / nsup;
    const int bn   = blockIdx.x - bm * nsup;
    const int n0   = bn << 6;

    // ---- stage B panel [K x 64] into LDS (16B per thread per pass) ----
    {
        const int tid = threadIdx.x;
        const int r0  = tid >> 3;                    // 16 rows per pass
        const int cb  = (tid & 7) << 4;              // byte column within row
#pragma unroll
        for (int rr = 0; rr < K; rr += 16) {
            const int r = r0 + rr;
            const char* gsrc = (const char*)(B + (size_t)r * N + n0) + cb;
            char*       ldst = (char*)(Bs + r * LDB) + cb;
#if USE_ASYNC_LDS
            __builtin_amdgcn_global_load_async_to_lds_b128(
                (__attribute__((address_space(1))) v4i_t*)(char*)gsrc,
                (__attribute__((address_space(3))) v4i_t*)ldst, 0, 0);
#else
            *(int4*)ldst = *(const int4*)gsrc;
#endif
        }
    }
#if USE_ASYNC_LDS
#if __has_builtin(__builtin_amdgcn_s_wait_asynccnt)
    __builtin_amdgcn_s_wait_asynccnt(0);
#else
    asm volatile("s_wait_asynccnt 0x0" ::: "memory");
#endif
#endif
    __syncthreads();

    const int wave = threadIdx.x >> 5;
    const int lane = threadIdx.x & 31;
    const int mt   = (bm << 2) + wave;               // wave-uniform m tile
    if (mt >= (M >> 4)) return;                      // uniform exit (no more barriers)
    const int hlf  = lane >> 4;
    const int mrow = (mt << 4) + (lane & 15);

    // ---- preload ALL A fragments for the K extent in one clause ----
    v8h alo[KS], ahi[KS];
    const _Float16* abase = A + (size_t)mrow * K + 8 * hlf;
#pragma unroll
    for (int s = 0; s < KS; ++s) {
        alo[s] = *(const v8h*)(abase + 32 * s);
        ahi[s] = *(const v8h*)(abase + 32 * s + 16);
    }

    v8f acc[4];
#pragma unroll
    for (int t = 0; t < 4; ++t)
#pragma unroll
        for (int j = 0; j < 8; ++j) acc[t][j] = 0.0f;

#pragma unroll
    for (int s = 0; s < KS; ++s) {
        v16h a;
#pragma unroll
        for (int i = 0; i < 8; ++i) { a[i] = alo[s][i]; a[8 + i] = ahi[s][i]; }
        const _Float16* brow = Bs + (32 * s + 8 * hlf) * LDB + (lane & 15);
        // load all 4 B fragments first (one ds clause), then 4 back-to-back WMMAs
        v16h bfrag[4];
#pragma unroll
        for (int t = 0; t < 4; ++t) {
#pragma unroll
            for (int i = 0; i < 8; ++i) { bfrag[t][i] = brow[i * LDB + 16 * t]; bfrag[t][8 + i] = brow[(16 + i) * LDB + 16 * t]; }
        }
#pragma unroll
        for (int t = 0; t < 4; ++t) {
            acc[t] = __builtin_amdgcn_wmma_f32_16x16x32_f16(
                         false, a, false, bfrag[t], (short)0, acc[t], false, false);
        }
    }

    // ---- store: D layout = VGPR j -> row m0 + j + 8*half, col n = lane&15 ----
#pragma unroll
    for (int t = 0; t < 4; ++t) {
        const int n  = n0 + 16 * t + (lane & 15);
        const float bv = bias ? bias[n] : 0.0f;
#pragma unroll
        for (int j = 0; j < 8; ++j) {
            const int m = (mt << 4) + j + 8 * hlf;
            const float v = acc[t][j] + bv;
            if (HALF_OUT) ((_Float16*)C)[(size_t)m * N + n] = (_Float16)v;
            else          ((float*)   C)[(size_t)m * N + n] = v;
        }
    }
}

// out[c][r] = (f16) in[r][c]     (in: rows x cols, f32)
__global__ void transpose_f32_to_f16(const float* __restrict__ in,
                                     _Float16* __restrict__ out,
                                     int rows, int cols)
{
    int i = blockIdx.x * blockDim.x + threadIdx.x;
    if (i >= rows * cols) return;
    int r = i / cols, c = i % cols;
    out[(size_t)c * rows + r] = (_Float16)in[i];
}

__global__ void zero_f32_kernel(float* __restrict__ p, int n)
{
    int i = blockIdx.x * blockDim.x + threadIdx.x;
    if (i < n) p[i] = 0.0f;
}

// h0 = relu(x @ lin0_W^T + b), write f32 + f16 copies
__global__ void lin0_kernel(const float* __restrict__ x,
                            const float* __restrict__ W,
                            const float* __restrict__ b,
                            float* __restrict__ h, _Float16* __restrict__ hh,
                            int N)
{
    int i = blockIdx.x * blockDim.x + threadIdx.x;
    if (i >= N * HD) return;
    int n = i >> 6, j = i & 63;
    const float* xr = x + (size_t)n * 15;
    const float* wr = W + (size_t)j * 15;
    float s = b[j];
#pragma unroll
    for (int k = 0; k < 15; ++k) s += xr[k] * wr[k];
    s = fmaxf(s, 0.0f);
    h[i] = s; hh[i] = (_Float16)s;
}

// ew = relu(edge_attr @ edge_W1^T + b1) -> f16 [E,128]
__global__ void edgehidden_kernel(const float* __restrict__ ea,
                                  const float* __restrict__ W1,
                                  const float* __restrict__ b1,
                                  _Float16* __restrict__ ewh, int E)
{
    int i = blockIdx.x * blockDim.x + threadIdx.x;
    if (i >= E * 128) return;
    int e = i >> 7, k = i & 127;
    const float* ar = ea + (size_t)e * 5;
    const float* wr = W1 + (size_t)k * 5;
    float s = b1[k];
#pragma unroll
    for (int j = 0; j < 5; ++j) s += ar[j] * wr[j];
    ewh[i] = (_Float16)fmaxf(s, 0.0f);
}

__global__ void deg_kernel(const int* __restrict__ dst, float* __restrict__ deg, int E)
{
    int i = blockIdx.x * blockDim.x + threadIdx.x;
    if (i < E) atomicAdd(&deg[dst[i]], 1.0f);
}
__global__ void invdeg_kernel(float* __restrict__ deg, int N)
{
    int i = blockIdx.x * blockDim.x + threadIdx.x;
    if (i < N) deg[i] = 1.0f / fmaxf(deg[i], 1.0f);
}

// wave-per-edge: msg[e,:] = h[src[e],:] @ W_e[e]  (f16 rows), atomic scatter-add
__global__ void msg_kernel(const float* __restrict__ h,
                           const _Float16* __restrict__ We,
                           const int* __restrict__ src,
                           const int* __restrict__ dst,
                           float* __restrict__ agg, int E)
{
    int gid = blockIdx.x * blockDim.x + threadIdx.x;
    int e = gid >> 5, lane = gid & 31;
    if (e >= E) return;                                  // wave-uniform
    int s = src[e], d = dst[e];
    float xa = h[(size_t)s * HD + lane];
    float xb = h[(size_t)s * HD + 32 + lane];
    const v2h* wrow = (const v2h*)(We + (size_t)e * (HD * HD));
    __builtin_prefetch(wrow + 32 * 32, 0, 1);            // second half of W_e[e]
    float a0 = 0.0f, a1 = 0.0f;
#pragma unroll 8
    for (int k = 0; k < HD; ++k) {
        float xk = (k < 32) ? __shfl(xa, k, 32) : __shfl(xb, k - 32, 32);
        v2h w = wrow[k * 32 + lane];                     // outputs 2*lane, 2*lane+1
        a0 += xk * (float)w[0];
        a1 += xk * (float)w[1];
    }
    atomicAdd(&agg[(size_t)d * HD + 2 * lane],     a0);
    atomicAdd(&agg[(size_t)d * HD + 2 * lane + 1], a1);
}

// m = f16( relu(agg * inv_deg + conv_b_i) )
__global__ void node_m_kernel(const float* __restrict__ agg,
                              const float* __restrict__ inv_deg,
                              const float* __restrict__ cb,
                              _Float16* __restrict__ mh, int N)
{
    int i = blockIdx.x * blockDim.x + threadIdx.x;
    if (i >= N * HD) return;
    int n = i >> 6, j = i & 63;
    float v = agg[i] * inv_deg[n] + cb[j];
    mh[i] = (_Float16)fmaxf(v, 0.0f);
}

__device__ __forceinline__ float sigm(float x) { return 1.0f / (1.0f + __expf(-x)); }

// GRU elementwise update (gi/gh already include biases from the GEMM)
__global__ void gru_kernel(const float* __restrict__ gi,
                           const float* __restrict__ gh,
                           float* __restrict__ h, _Float16* __restrict__ hh, int N)
{
    int i = blockIdx.x * blockDim.x + threadIdx.x;
    if (i >= N * HD) return;
    int n = i >> 6, j = i & 63;
    const float* gir = gi + (size_t)n * 192;
    const float* ghr = gh + (size_t)n * 192;
    float r  = sigm(gir[j]       + ghr[j]);
    float z  = sigm(gir[64 + j]  + ghr[64 + j]);
    float nn = tanhf(gir[128 + j] + r * ghr[128 + j]);
    float hv = (1.0f - z) * nn + z * h[i];
    h[i] = hv; hh[i] = (_Float16)hv;
}

// ---- Set2Set --------------------------------------------------------------
__device__ __forceinline__ unsigned enc_ord(float f) {
    unsigned u = __float_as_uint(f);
    return (u & 0x80000000u) ? ~u : (u | 0x80000000u);
}
__device__ __forceinline__ float dec_ord(unsigned u) {
    unsigned v = (u & 0x80000000u) ? (u & 0x7FFFFFFFu) : ~u;
    return __uint_as_float(v);
}

__global__ void lstm_kernel(const float* __restrict__ q_star,
                            const float* __restrict__ Wih,  // [256,128]
                            const float* __restrict__ Whh,  // [256,64]
                            const float* __restrict__ bih,
                            const float* __restrict__ bhh,
                            const float* __restrict__ hl_in,
                            float* __restrict__ cl,
                            float* __restrict__ hl_out, int G)
{
    int i = blockIdx.x * blockDim.x + threadIdx.x;
    if (i >= G * HD) return;
    int g = i >> 6, j = i & 63;
    const float* qs = q_star + (size_t)g * 128;
    const float* hr = hl_in  + (size_t)g * HD;
    float gate[4];
#pragma unroll
    for (int t = 0; t < 4; ++t) {
        int row = t * HD + j;
        float s = bih[row] + bhh[row];
        const float* wi = Wih + (size_t)row * 128;
        for (int k = 0; k < 128; ++k) s += qs[k] * wi[k];
        const float* wh = Whh + (size_t)row * HD;
        for (int k = 0; k < HD; ++k) s += hr[k] * wh[k];
        gate[t] = s;
    }
    float c = sigm(gate[1]) * cl[i] + sigm(gate[0]) * tanhf(gate[2]);
    cl[i] = c;
    hl_out[i] = sigm(gate[3]) * tanhf(c);
}

// q_star[:, :64] = q (=hl) ; q_star[:, 64:] = 0 ; init segment max/denoms
__global__ void qstar_q_kernel(const float* __restrict__ hl,
                               float* __restrict__ q_star,
                               unsigned* __restrict__ emax_u,
                               float* __restrict__ denom, int G)
{
    int i = blockIdx.x * blockDim.x + threadIdx.x;
    if (i >= G * HD) return;
    int g = i >> 6, j = i & 63;
    q_star[(size_t)g * 128 + j]      = hl[i];
    q_star[(size_t)g * 128 + 64 + j] = 0.0f;
    if (j == 0) { emax_u[g] = enc_ord(-__builtin_inff()); denom[g] = 0.0f; }
}

__global__ void att_e_kernel(const float* __restrict__ h,
                             const float* __restrict__ q,     // hl
                             const int* __restrict__ batch,
                             float* __restrict__ ebuf,
                             unsigned* __restrict__ emax_u, int N)
{
    int n = blockIdx.x * blockDim.x + threadIdx.x;
    if (n >= N) return;
    int b = batch[n];
    const float* hr = h + (size_t)n * HD;
    const float* qr = q + (size_t)b * HD;
    float s = 0.0f;
#pragma unroll 8
    for (int k = 0; k < HD; ++k) s += hr[k] * qr[k];
    ebuf[n] = s;
    atomicMax(&emax_u[b], enc_ord(s));
}

__global__ void att_exp_kernel(const float* __restrict__ ebuf,
                               const int* __restrict__ batch,
                               const unsigned* __restrict__ emax_u,
                               float* __restrict__ eexp,
                               float* __restrict__ denom, int N)
{
    int n = blockIdx.x * blockDim.x + threadIdx.x;
    if (n >= N) return;
    int b = batch[n];
    float v = __expf(ebuf[n] - dec_ord(emax_u[b]));
    eexp[n] = v;
    atomicAdd(&denom[b], v);
}

// q_star[:, 64:] += a[n] * h[n]   (segment softmax-weighted sum)
__global__ void att_r_kernel(const float* __restrict__ h,
                             const int* __restrict__ batch,
                             const float* __restrict__ eexp,
                             const float* __restrict__ denom,
                             float* __restrict__ q_star, int N)
{
    int i = blockIdx.x * blockDim.x + threadIdx.x;
    if (i >= N * HD) return;
    int n = i >> 6, j = i & 63;
    int b = batch[n];
    float a = eexp[n] / fmaxf(denom[b], 1e-16f);
    atomicAdd(&q_star[(size_t)b * 128 + 64 + j], a * h[i]);
}

__global__ void lin1_kernel(const float* __restrict__ q_star,
                            const float* __restrict__ W,    // [64,128]
                            const float* __restrict__ b,
                            float* __restrict__ y, int G)
{
    int i = blockIdx.x * blockDim.x + threadIdx.x;
    if (i >= G * HD) return;
    int g = i >> 6, j = i & 63;
    const float* qs = q_star + (size_t)g * 128;
    const float* wr = W + (size_t)j * 128;
    float s = b[j];
    for (int k = 0; k < 128; ++k) s += qs[k] * wr[k];
    y[i] = fmaxf(s, 0.0f);
}

__global__ void lin2_kernel(const float* __restrict__ y,
                            const float* __restrict__ W,    // [12,64]
                            const float* __restrict__ b,
                            float* __restrict__ out, int G)
{
    int i = blockIdx.x * blockDim.x + threadIdx.x;
    if (i >= G * 12) return;
    int g = i / 12, o = i % 12;
    const float* yr = y + (size_t)g * HD;
    const float* wr = W + (size_t)o * HD;
    float s = b[o];
    for (int k = 0; k < HD; ++k) s += yr[k] * wr[k];
    out[i] = s;
}

// ---------------------------------------------------------------------------
static inline int cdiv(long long a, long long b) { return (int)((a + b - 1) / b); }

extern "C" void kernel_launch(void* const* d_in, const int* in_sizes, int n_in,
                              void* d_out, int out_size, void* d_ws, size_t ws_size,
                              hipStream_t stream)
{
    (void)n_in; (void)ws_size;
    const float* x         = (const float*)d_in[0];
    const float* edge_attr = (const float*)d_in[1];
    const int*   edge_idx  = (const int*)  d_in[2];
    const int*   batch     = (const int*)  d_in[3];
    // d_in[4] = num_graphs (device scalar); G recovered from out_size
    const float* lin0_W   = (const float*)d_in[5];
    const float* lin0_b   = (const float*)d_in[6];
    const float* edge_W1  = (const float*)d_in[7];
    const float* edge_b1  = (const float*)d_in[8];
    const float* edge_W2  = (const float*)d_in[9];   // [4096,128]
    const float* edge_b2  = (const float*)d_in[10];  // [4096]
    const float* conv_b   = (const float*)d_in[11];  // [6,64]
    const float* gru_Wih  = (const float*)d_in[12];  // [192,64]
    const float* gru_Whh  = (const float*)d_in[13];  // [192,64]
    const float* gru_bih  = (const float*)d_in[14];
    const float* gru_bhh  = (const float*)d_in[15];
    const float* lstm_Wih = (const float*)d_in[16];  // [256,128]
    const float* lstm_Whh = (const float*)d_in[17];  // [256,64]
    const float* lstm_bih = (const float*)d_in[18];
    const float* lstm_bhh = (const float*)d_in[19];
    const float* lin1_W   = (const float*)d_in[20];  // [64,128]
    const float* lin1_b   = (const float*)d_in[21];
    const float* lin2_W   = (const float*)d_in[22];  // [12,64]
    const float* lin2_b   = (const float*)d_in[23];

    const int N = in_sizes[0] / 15;
    const int E = in_sizes[1] / 5;
    const int G = out_size / 12;
    const int* src = edge_idx;
    const int* dst = edge_idx + E;

    // --- workspace carve (256B aligned) ---
    char* p = (char*)d_ws;
    auto carve = [&](size_t bytes) -> char* {
        char* r = p; p += (bytes + 255) & ~(size_t)255; return r;
    };
    _Float16* We_h   = (_Float16*)carve((size_t)E * 4096 * 2);  // 134 MB, L2-resident
    _Float16* ew_h   = (_Float16*)carve((size_t)E * 128 * 2);
    _Float16* W2T_h  = (_Float16*)carve((size_t)128 * 4096 * 2);
    _Float16* WihT_h = (_Float16*)carve((size_t)64 * 192 * 2);
    _Float16* WhhT_h = (_Float16*)carve((size_t)64 * 192 * 2);
    float*    h_f    = (float*)   carve((size_t)N * HD * 4);
    _Float16* h_h    = (_Float16*)carve((size_t)N * HD * 2);
    _Float16* m_h    = (_Float16*)carve((size_t)N * HD * 2);
    float*    agg    = (float*)   carve((size_t)N * HD * 4);
    float*    gi     = (float*)   carve((size_t)N * 192 * 4);
    float*    gh     = (float*)   carve((size_t)N * 192 * 4);
    float*    invdeg = (float*)   carve((size_t)N * 4);
    float*    q_star = (float*)   carve((size_t)G * 128 * 4);
    float*    hlA    = (float*)   carve((size_t)G * HD * 4);
    float*    hlB    = (float*)   carve((size_t)G * HD * 4);
    float*    cl     = (float*)   carve((size_t)G * HD * 4);
    float*    ebuf   = (float*)   carve((size_t)N * 4);
    float*    eexp   = (float*)   carve((size_t)N * 4);
    unsigned* emax_u = (unsigned*)carve((size_t)G * 4);
    float*    denom  = (float*)   carve((size_t)G * 4);
    float*    ybuf   = (float*)   carve((size_t)G * HD * 4);

    const int T = 256;

    // --- weight repacks (f32 -> f16, transposed to [K,N]) ---
    transpose_f32_to_f16<<<cdiv((size_t)4096 * 128, T), T, 0, stream>>>(edge_W2, W2T_h, 4096, 128);
    transpose_f32_to_f16<<<cdiv(192 * 64, T), T, 0, stream>>>(gru_Wih, WihT_h, 192, 64);
    transpose_f32_to_f16<<<cdiv(192 * 64, T), T, 0, stream>>>(gru_Whh, WhhT_h, 192, 64);

    // --- input embeddings ---
    lin0_kernel<<<cdiv((size_t)N * HD, T), T, 0, stream>>>(x, lin0_W, lin0_b, h_f, h_h, N);
    edgehidden_kernel<<<cdiv((size_t)E * 128, T), T, 0, stream>>>(edge_attr, edge_W1, edge_b1, ew_h, E);

    // --- big WMMA GEMM: W_e = ew @ W2^T + b2, f16 out (M=E, N=4096, K=128) ---
    wmma_gemm_kernel<128, true><<<(E / 64) * (4096 / 64), 128, 0, stream>>>(
        ew_h, W2T_h, edge_b2, (void*)We_h, E, 4096);

    // --- scatter-mean denominator ---
    zero_f32_kernel<<<cdiv(N, T), T, 0, stream>>>(invdeg, N);
    deg_kernel<<<cdiv(E, T), T, 0, stream>>>(dst, invdeg, E);
    invdeg_kernel<<<cdiv(N, T), T, 0, stream>>>(invdeg, N);

    // --- 6 message-passing + GRU steps (W_e stays hot in L2) ---
    for (int step = 0; step < 6; ++step) {
        zero_f32_kernel<<<cdiv((size_t)N * HD, T), T, 0, stream>>>(agg, N * HD);
        msg_kernel<<<cdiv((size_t)E * 32, T), T, 0, stream>>>(h_f, We_h, src, dst, agg, E);
        node_m_kernel<<<cdiv((size_t)N * HD, T), T, 0, stream>>>(agg, invdeg, conv_b + step * HD, m_h, N);
        wmma_gemm_kernel<64, false><<<(N / 64) * (192 / 64), 128, 0, stream>>>(
            m_h, WihT_h, gru_bih, (void*)gi, N, 192);
        wmma_gemm_kernel<64, false><<<(N / 64) * (192 / 64), 128, 0, stream>>>(
            h_h, WhhT_h, gru_bhh, (void*)gh, N, 192);
        gru_kernel<<<cdiv((size_t)N * HD, T), T, 0, stream>>>(gi, gh, h_f, h_h, N);
    }

    // --- Set2Set readout ---
    zero_f32_kernel<<<cdiv((size_t)G * 128, T), T, 0, stream>>>(q_star, G * 128);
    zero_f32_kernel<<<cdiv((size_t)G * HD, T), T, 0, stream>>>(hlA, G * HD);
    zero_f32_kernel<<<cdiv((size_t)G * HD, T), T, 0, stream>>>(cl, G * HD);
    float* hl_in = hlA; float* hl_out = hlB;
    for (int it = 0; it < 6; ++it) {
        lstm_kernel<<<cdiv((size_t)G * HD, T), T, 0, stream>>>(
            q_star, lstm_Wih, lstm_Whh, lstm_bih, lstm_bhh, hl_in, cl, hl_out, G);
        qstar_q_kernel<<<cdiv((size_t)G * HD, T), T, 0, stream>>>(hl_out, q_star, emax_u, denom, G);
        att_e_kernel<<<cdiv(N, T), T, 0, stream>>>(h_f, hl_out, batch, ebuf, emax_u, N);
        att_exp_kernel<<<cdiv(N, T), T, 0, stream>>>(ebuf, batch, emax_u, eexp, denom, N);
        att_r_kernel<<<cdiv((size_t)N * HD, T), T, 0, stream>>>(h_f, batch, eexp, denom, q_star, N);
        float* tmp = hl_in; hl_in = hl_out; hl_out = tmp;
    }

    // --- output head ---
    lin1_kernel<<<cdiv((size_t)G * HD, T), T, 0, stream>>>(q_star, lin1_W, lin1_b, ybuf, G);
    lin2_kernel<<<cdiv((size_t)G * 12, T), T, 0, stream>>>(ybuf, lin2_W, lin2_b, (float*)d_out, G);
}
